// MultiHeadAttention_45509473468668
// MI455X (gfx1250) — matrix operand order repaired and checked by reading, not compiled, not run on hardware
//
#include <hip/hip_runtime.h>
#include <hip/hip_bf16.h>

// ---------------------------------------------------------------------------
// CDNA5 (gfx1250) multi-head attention forward:
//   qkv = x @ w_attn + b_attn ; RoPE(q,k) ; causal flash attention ; proj
// bf16 WMMA (v_wmma_f32_16x16x32_bf16) + async global->LDS tile staging.
// ---------------------------------------------------------------------------

#ifndef __has_builtin
#define __has_builtin(x) 0
#endif
#if __has_builtin(__builtin_amdgcn_global_load_async_to_lds_b128) && \
    __has_builtin(__builtin_amdgcn_s_wait_asynccnt)
#define MHA_ASYNC 1
#else
#define MHA_ASYNC 0
#endif

typedef __bf16 bf16_t;
typedef bf16_t v16bf __attribute__((ext_vector_type(16)));
typedef bf16_t v8bf  __attribute__((ext_vector_type(8)));
typedef float  v8f   __attribute__((ext_vector_type(8)));
typedef int    v4i   __attribute__((ext_vector_type(4)));

static __device__ __forceinline__ unsigned short f2bf(float x) {
    return __builtin_bit_cast(unsigned short, (__bf16)x);
}

// 16B global -> LDS copy. Async path uses the gfx1250 ASYNCcnt-tracked direct
// copy (no VGPR bounce); fallback is a register-staged vector copy.
static __device__ __forceinline__ void cp_b128(const unsigned short* g,
                                               unsigned short* l) {
#if MHA_ASYNC
    __builtin_amdgcn_global_load_async_to_lds_b128(
        (__attribute__((address_space(1))) v4i*)g,
        (__attribute__((address_space(3))) v4i*)l, 0, 0);
#else
    *reinterpret_cast<v8bf*>(l) = *reinterpret_cast<const v8bf*>(g);
#endif
}
static __device__ __forceinline__ void wait_cp() {
#if MHA_ASYNC
    __builtin_amdgcn_s_wait_asynccnt(0);
#endif
}

static __device__ __forceinline__ v16bf mk16(v8bf lo, v8bf hi) {
    v16bf r;
#pragma unroll
    for (int i = 0; i < 8; ++i) { r[i] = lo[i]; r[i + 8] = hi[i]; }
    return r;
}

// A-matrix 16x32 bf16 fragment (ISA 7.12.2): lane row = lane&15,
// K = {half*8 + 0..7, 16 + half*8 + 0..7}, half = lane>>4.
static __device__ __forceinline__ v16bf load_frag_A(const unsigned short* base,
                                                    int ld, int lane) {
    int r = lane & 15, half = lane >> 4;
    const unsigned short* p = base + (size_t)r * ld;
    v8bf lo = *reinterpret_cast<const v8bf*>(p + half * 8);
    v8bf hi = *reinterpret_cast<const v8bf*>(p + 16 + half * 8);
    return mk16(lo, hi);
}

// B-matrix 32x16 bf16 fragment: lane column = lane&15, K = {half*8 + 0..7, 16 + half*8 + 0..7}.
// base points at a K-contiguous, n-major tile with leading dim ld.
static __device__ __forceinline__ v16bf load_frag_B(const unsigned short* base,
                                                    int ld, int lane) {
    int n = lane & 15, half = lane >> 4;
    const unsigned short* p = base + (size_t)n * ld + half * 8;
    v8bf lo = *reinterpret_cast<const v8bf*>(p);
    v8bf hi = *reinterpret_cast<const v8bf*>(p + 16);
    return mk16(lo, hi);
}

static __device__ __forceinline__ v8f wmma_bf16(v16bf a, v16bf b, v8f c) {
    return __builtin_amdgcn_wmma_f32_16x16x32_bf16(false, a, false, b,
                                                   (short)0, c, false, false);
}

static __device__ __forceinline__ v8f v8f_zero() {
    v8f z;
#pragma unroll
    for (int i = 0; i < 8; ++i) z[i] = 0.0f;
    return z;
}

// ---------------------------------------------------------------------------
// f32 -> bf16 elementwise convert
// ---------------------------------------------------------------------------
__global__ void mha_cvt_bf16(const float* __restrict__ src,
                             unsigned short* __restrict__ dst, int n) {
    int i = blockIdx.x * blockDim.x + threadIdx.x;
    if (i < n) dst[i] = f2bf(src[i]);
}

// f32 W[K,N] -> bf16 W^T[N,K] (K-contiguous rows for vectorized LDS staging)
__global__ void mha_cvt_transpose_bf16(const float* __restrict__ W,
                                       unsigned short* __restrict__ Wt,
                                       int K, int N) {
    int i = blockIdx.x * blockDim.x + threadIdx.x;
    if (i >= K * N) return;
    int k = i % K, n = i / K;
    Wt[(size_t)n * K + k] = f2bf(W[(size_t)k * N + n]);
}

// ---------------------------------------------------------------------------
// GEMM: C[M,N] f32 = A[M,K]bf16 @ (Bt[N,K]bf16)^T + bias[N]
// Block tile 64x128, 4 waves (2x2), wave tile 32x64, K-step 32,
// double-buffered async LDS staging.
// ---------------------------------------------------------------------------
__global__ __launch_bounds__(128) void mha_gemm_bf16(
    const unsigned short* __restrict__ A, const unsigned short* __restrict__ Bt,
    const float* __restrict__ bias, float* __restrict__ Cout,
    int M, int N, int K) {
    __shared__ unsigned short As[2][64][40];    // [m][k], padded
    __shared__ unsigned short Bs[2][128][40];   // [n][k], padded

    const int tid = threadIdx.x;
    const int lane = tid & 31;
    const int w = tid >> 5;
    const int wm = w >> 1, wn = w & 1;
    const int m0 = blockIdx.y * 64;
    const int n0 = blockIdx.x * 128;

    v8f acc[2][4];
#pragma unroll
    for (int i = 0; i < 2; ++i)
#pragma unroll
        for (int j = 0; j < 4; ++j) acc[i][j] = v8f_zero();

    auto issue = [&](int k0, int buf) {
        for (int c = tid; c < 256; c += 128) {          // A: 64x32
            int row = c >> 2, kc = (c & 3) << 3;
            cp_b128(&A[(size_t)(m0 + row) * K + k0 + kc], &As[buf][row][kc]);
        }
        for (int c = tid; c < 512; c += 128) {          // B: 128x32
            int n = c >> 2, kc = (c & 3) << 3;
            cp_b128(&Bt[(size_t)(n0 + n) * K + k0 + kc], &Bs[buf][n][kc]);
        }
    };

    const int nk = K >> 5;
    issue(0, 0);
    for (int it = 0; it < nk; ++it) {
        const int buf = it & 1;
        wait_cp();
        __syncthreads();
        if (it + 1 < nk) issue((it + 1) << 5, buf ^ 1);

        v16bf a0 = load_frag_A(&As[buf][wm * 32][0], 40, lane);
        v16bf a1 = load_frag_A(&As[buf][wm * 32 + 16][0], 40, lane);
#pragma unroll
        for (int cs = 0; cs < 4; ++cs) {
            v16bf b = load_frag_B(&Bs[buf][wn * 64 + cs * 16][0], 40, lane);
            acc[0][cs] = wmma_bf16(a0, b, acc[0][cs]);
            acc[1][cs] = wmma_bf16(a1, b, acc[1][cs]);
        }
        __syncthreads();   // frag reads done before next overwrite of buf^1
    }

    const int r = lane & 15, half = lane >> 4;
#pragma unroll
    for (int rs = 0; rs < 2; ++rs)
#pragma unroll
        for (int cs = 0; cs < 4; ++cs)
#pragma unroll
            for (int v = 0; v < 8; ++v) {
                int row = m0 + wm * 32 + rs * 16 + v + half * 8;
                int col = n0 + wn * 64 + cs * 16 + r;
                Cout[(size_t)row * N + col] = acc[rs][cs][v] + bias[col];
            }
}

// ---------------------------------------------------------------------------
// RoPE + split heads: qkv f32 [B,T,3C] -> Q/K bf16 [B,H,T,64], V bf16 [B,H,64,T]
// (V pre-transposed so flash can stage it with contiguous 16B copies)
// ---------------------------------------------------------------------------
__global__ void mha_rope_split(const float* __restrict__ qkv,
                               unsigned short* __restrict__ Qb,
                               unsigned short* __restrict__ Kb,
                               unsigned short* __restrict__ Vt,
                               int B, int T, int H) {
    int idx = blockIdx.x * blockDim.x + threadIdx.x;
    int total = B * H * T * 32;
    if (idx >= total) return;
    int d = idx & 31;
    int t = (idx >> 5) % T;
    int h = (idx / (32 * T)) % H;
    int b = idx / (32 * T * H);
    int C = H * 64;
    int bh = b * H + h;
    size_t base = ((size_t)(b * T + t)) * (3 * C) + h * 64;

    float q1 = qkv[base + d],         q2 = qkv[base + d + 32];
    float k1 = qkv[base + C + d],     k2 = qkv[base + C + d + 32];
    float v1 = qkv[base + 2 * C + d], v2 = qkv[base + 2 * C + d + 32];

    float freq = __powf(10000.0f, -(float)(2 * d) / 64.0f);
    float s, c;
    __sincosf((float)t * freq, &s, &c);

    size_t o = ((size_t)bh * T + t) * 64 + d;
    Qb[o]      = f2bf(q1 * c - q2 * s);
    Qb[o + 32] = f2bf(q2 * c + q1 * s);
    Kb[o]      = f2bf(k1 * c - k2 * s);
    Kb[o + 32] = f2bf(k2 * c + k1 * s);
    Vt[((size_t)(bh * 64 + d)) * T + t]      = f2bf(v1);
    Vt[((size_t)(bh * 64 + d + 32)) * T + t] = f2bf(v2);
}

// ---------------------------------------------------------------------------
// Flash attention (causal, online softmax). Block = 4 waves = 64 q rows.
// Each wave: 16 q rows, acc 16x64 f32. kv streamed in 32-row double-buffered
// async LDS tiles.
// ---------------------------------------------------------------------------
__global__ __launch_bounds__(128) void mha_flash_attn(
    const unsigned short* __restrict__ Q, const unsigned short* __restrict__ Kt,
    const unsigned short* __restrict__ Vt, unsigned short* __restrict__ Y,
    int T, int H) {
    __shared__ unsigned short Ks[2][32][72];   // [kv][d] row-major
    __shared__ unsigned short Vs[2][64][40];   // [d][kv]
    __shared__ unsigned short Pw[4][16][40];   // per-wave P relayout scratch

    const int tid = threadIdx.x, lane = tid & 31, w = tid >> 5;
    const int r = lane & 15, half = lane >> 4;
    const int qtile = blockIdx.x, bh = blockIdx.y;
    const int b = bh / H, h = bh % H;

    const unsigned short* qg = Q + (size_t)bh * T * 64;
    const unsigned short* kg = Kt + (size_t)bh * T * 64;
    const unsigned short* vg = Vt + (size_t)bh * 64 * T;

    const int qrow0 = qtile * 64 + w * 16;
    // Q fragments held for the whole kv loop (A layout, K = d in two 32-chunks)
    v16bf qa0 = load_frag_A(qg + (size_t)qrow0 * 64 + 0, 64, lane);
    v16bf qa1 = load_frag_A(qg + (size_t)qrow0 * 64 + 32, 64, lane);

    v8f acc[4];
#pragma unroll
    for (int i = 0; i < 4; ++i) acc[i] = v8f_zero();
    float m_i[8], l_i[8];
#pragma unroll
    for (int v = 0; v < 8; ++v) { m_i[v] = -3.0e38f; l_i[v] = 0.0f; }

    auto issue_kv = [&](int j, int buf) {
        for (int c = tid; c < 256; c += 128) {          // K: 32 kv x 64 d
            int kr = c >> 3, dc = (c & 7) << 3;
            cp_b128(kg + ((size_t)(j * 32 + kr)) * 64 + dc, &Ks[buf][kr][dc]);
        }
        for (int c = tid; c < 256; c += 128) {          // V^T: 64 d x 32 kv
            int dr = c >> 2, kvc = (c & 3) << 3;
            cp_b128(vg + (size_t)dr * T + j * 32 + kvc, &Vs[buf][dr][kvc]);
        }
    };

    const int jmax_w = (qrow0 + 15) >> 5;      // last kv block this wave needs
    const int jmax = 2 * qtile + 1;            // block-wide loop bound

    issue_kv(0, 0);
    for (int j = 0; j <= jmax; ++j) {
        const int buf = j & 1;
        wait_cp();
        __syncthreads();
        if (j < jmax) issue_kv(j + 1, buf ^ 1);

        const bool active = (j <= jmax_w);     // wave-uniform: EXEC stays full
        v8f p0, p1;
        if (active) {
            // S = Q @ K^T for two 16-col kv subtiles, K-dim = 64 (chained WMMA)
            v8f s0 = wmma_bf16(qa0, load_frag_B(&Ks[buf][0][0], 72, lane), v8f_zero());
            s0 = wmma_bf16(qa1, load_frag_B(&Ks[buf][0][32], 72, lane), s0);
            v8f s1 = wmma_bf16(qa0, load_frag_B(&Ks[buf][16][0], 72, lane), v8f_zero());
            s1 = wmma_bf16(qa1, load_frag_B(&Ks[buf][16][32], 72, lane), s1);

            float alpha[8];
#pragma unroll
            for (int v = 0; v < 8; ++v) {
                int qr = qrow0 + v + half * 8;
                float a0 = s0[v] * 0.125f;
                float a1 = s1[v] * 0.125f;
                if (j * 32 + r > qr)      a0 = -3.0e38f;
                if (j * 32 + 16 + r > qr) a1 = -3.0e38f;
                float rmax = fmaxf(a0, a1);
#pragma unroll
                for (int off = 1; off < 16; off <<= 1)
                    rmax = fmaxf(rmax, __shfl_xor(rmax, off, 32));
                float mn = fmaxf(m_i[v], rmax);
                float al = __expf(m_i[v] - mn);
                float e0 = __expf(a0 - mn);
                float e1 = __expf(a1 - mn);
                float rs = e0 + e1;
#pragma unroll
                for (int off = 1; off < 16; off <<= 1)
                    rs += __shfl_xor(rs, off, 32);
                l_i[v] = l_i[v] * al + rs;
                m_i[v] = mn;
                alpha[v] = al;
                p0[v] = e0;
                p1[v] = e1;
            }
#pragma unroll
            for (int cs = 0; cs < 4; ++cs)
#pragma unroll
                for (int v = 0; v < 8; ++v) acc[cs][v] *= alpha[v];
            // write P (C layout) into per-wave LDS for A-fragment reload
#pragma unroll
            for (int v = 0; v < 8; ++v) {
                Pw[w][v + half * 8][r]      = f2bf(p0[v]);
                Pw[w][v + half * 8][16 + r] = f2bf(p1[v]);
            }
        }
        __syncthreads();
        if (active) {
            v16bf pa = load_frag_A(&Pw[w][0][0], 40, lane);
#pragma unroll
            for (int cs = 0; cs < 4; ++cs) {
                v16bf vf = load_frag_B(&Vs[buf][cs * 16][0], 40, lane);
                acc[cs] = wmma_bf16(pa, vf, acc[cs]);
            }
        }
    }

    // epilogue: normalize and write y as bf16 [B,T,C] for the proj GEMM
    const int C = H * 64;
#pragma unroll
    for (int v = 0; v < 8; ++v) {
        float inv = 1.0f / l_i[v];
        int t = qrow0 + v + half * 8;
#pragma unroll
        for (int cs = 0; cs < 4; ++cs) {
            int d = cs * 16 + r;
            Y[((size_t)(b * T + t)) * C + h * 64 + d] = f2bf(acc[cs][v] * inv);
        }
    }
}

// ---------------------------------------------------------------------------
// Host-side launch
// ---------------------------------------------------------------------------
extern "C" void kernel_launch(void* const* d_in, const int* in_sizes, int n_in,
                              void* d_out, int out_size, void* d_ws, size_t ws_size,
                              hipStream_t stream) {
    (void)in_sizes; (void)n_in; (void)out_size; (void)ws_size;
    const float* x      = (const float*)d_in[0];
    const float* w_attn = (const float*)d_in[1];
    const float* b_attn = (const float*)d_in[2];
    const float* w_proj = (const float*)d_in[3];
    const float* b_proj = (const float*)d_in[4];
    float* out = (float*)d_out;

    const int B = 2, T = 2048, C = 1024, H = 16;
    const int M = B * T;           // 4096
    const int C3 = 3 * C;          // 3072

    char* ws = (char*)d_ws;
    size_t off = 0;
    auto alloc = [&](size_t bytes) {
        void* p = ws + off;
        off += (bytes + 255) & ~(size_t)255;
        return p;
    };
    unsigned short* x_bf  = (unsigned short*)alloc((size_t)M * C * 2);
    unsigned short* wa_t  = (unsigned short*)alloc((size_t)C * C3 * 2);  // [3C][C]
    unsigned short* wp_t  = (unsigned short*)alloc((size_t)C * C * 2);   // [C][C]
    float*          qkv   = (float*)alloc((size_t)M * C3 * 4);
    unsigned short* Qb    = (unsigned short*)alloc((size_t)M * C * 2);
    unsigned short* Kb    = (unsigned short*)alloc((size_t)M * C * 2);
    unsigned short* Vt    = (unsigned short*)alloc((size_t)M * C * 2);
    unsigned short* Yb    = (unsigned short*)alloc((size_t)M * C * 2);

    // 1) convert x; convert+transpose weights
    {
        int n = M * C;
        mha_cvt_bf16<<<(n + 255) / 256, 256, 0, stream>>>(x, x_bf, n);
        n = C * C3;
        mha_cvt_transpose_bf16<<<(n + 255) / 256, 256, 0, stream>>>(w_attn, wa_t, C, C3);
        n = C * C;
        mha_cvt_transpose_bf16<<<(n + 255) / 256, 256, 0, stream>>>(w_proj, wp_t, C, C);
    }
    // 2) qkv = x @ w_attn + b_attn
    mha_gemm_bf16<<<dim3(C3 / 128, M / 64), 128, 0, stream>>>(
        x_bf, wa_t, b_attn, qkv, M, C3, C);
    // 3) RoPE + head split -> Q/K bf16 [B,H,T,64], V bf16 [B,H,64,T]
    {
        int n = B * H * T * 32;
        mha_rope_split<<<(n + 255) / 256, 256, 0, stream>>>(qkv, Qb, Kb, Vt, B, T, H);
    }
    // 4) causal flash attention -> Yb bf16 [B,T,C]
    mha_flash_attn<<<dim3(T / 64, B * H), 128, 0, stream>>>(Qb, Kb, Vt, Yb, T, H);
    // 5) out = y @ w_proj + b_proj
    mha_gemm_bf16<<<dim3(C / 128, M / 64), 128, 0, stream>>>(
        Yb, wp_t, b_proj, out, M, C, C);
}
